// DynamicGraphNeuralNetwork_14276471292829
// MI455X (gfx1250) — matrix operand order repaired and checked by reading, not compiled, not run on hardware
//
#include <hip/hip_runtime.h>
#include <hip/hip_bf16.h>

// ---------------------------------------------------------------------------
// Types / helpers
// ---------------------------------------------------------------------------
typedef __bf16 bf16_t;
typedef bf16_t v16bf __attribute__((ext_vector_type(16)));
typedef float  v8f   __attribute__((ext_vector_type(8)));

#define DEV __device__ __forceinline__

DEV bf16_t f2bf(float f) {
  unsigned u = __builtin_bit_cast(unsigned, f);
  u += 0x7FFFu + ((u >> 16) & 1u);            // round-to-nearest-even
  unsigned short h = (unsigned short)(u >> 16);
  return __builtin_bit_cast(bf16_t, h);
}

DEV void atomAddF(float* p, float v) { unsafeAtomicAdd(p, v); }

// Async global -> LDS copy, 16 bytes per lane (GLOBAL_LOAD_ASYNC_TO_LDS_B128,
// tracked by ASYNCcnt). LDS dest VGPR holds the wave-relative LDS byte offset,
// which is the low 32 bits of the generic pointer to a __shared__ object.
DEV void async_g2l_b128(void* lds_dst, const void* gsrc) {
  unsigned lds_off = (unsigned)(unsigned long long)lds_dst;
  asm volatile("global_load_async_to_lds_b128 %0, %1, off"
               :: "v"(lds_off), "v"(gsrc)
               : "memory");
}
DEV void wait_asynccnt0() {
  asm volatile("s_wait_asynccnt 0" ::: "memory");
}

// ---------------------------------------------------------------------------
// Constants of the problem
// ---------------------------------------------------------------------------
#define NB    16          // graphs
#define NPG0  4096        // nodes per graph at input
#define NT    8           // timesteps
#define F_IN  100
#define EDGES 524288      // edges per timestep (all graphs)
#define OUTF  256
#define KPAD  128         // padded K for every layer (100->128, 128->128)

// ---------------------------------------------------------------------------
// Fill kernels
// ---------------------------------------------------------------------------
__global__ void k_fill_f32(float* p, float v, long n) {
  long i = (long)blockIdx.x * blockDim.x + threadIdx.x;
  if (i < n) p[i] = v;
}
__global__ void k_fill_i32(int* p, int v, long n) {
  long i = (long)blockIdx.x * blockDim.x + threadIdx.x;
  if (i < n) p[i] = v;
}

// ---------------------------------------------------------------------------
// Weight conversion: f32 [K x N] -> bf16 [KPAD x N] (zero padded K rows)
// ---------------------------------------------------------------------------
__global__ void k_conv_w(const float* __restrict__ W, bf16_t* __restrict__ Wb,
                         int K, int Kpad, int N) {
  int i = blockIdx.x * blockDim.x + threadIdx.x;
  if (i >= Kpad * N) return;
  int k = i / N, n = i % N;
  Wb[i] = f2bf(k < K ? W[(size_t)k * N + n] : 0.0f);
}

// ---------------------------------------------------------------------------
// GEMM: Y[M x N] = X[M x K] (f32, row stride ldx) @ Wb[KPAD x N] (bf16)
// block = 256 threads = 8 waves -> 128 rows x 128 cols per block
// Weight chunk (KPAD x 128 bf16 = 32 KB) async-staged into LDS once per block.
// grid = (M/128, N/128)
// ---------------------------------------------------------------------------
__global__ void __launch_bounds__(256) k_gemm_bf16(
    const float* __restrict__ X, int ldx, int K,
    const bf16_t* __restrict__ Wb, int N,
    float* __restrict__ Y) {
  __shared__ bf16_t sW[KPAD * 128];           // [k][col], 32 KB

  const int lane = threadIdx.x & 31;
  const int wave = threadIdx.x >> 5;
  const int half = lane >> 4;
  const int l15  = lane & 15;
  const int colBase = blockIdx.y * 128;

  // ---- stage weight chunk into LDS via async copies (8 x B128 per thread) --
  {
    const bf16_t* gW = Wb + colBase;          // row stride = N
    for (int idx = threadIdx.x; idx < KPAD * 16; idx += 256) {
      int r  = idx >> 4;                      // weight row (K index)
      int c8 = (idx & 15) * 8;                // 8 bf16 = 16 bytes per transfer
      async_g2l_b128(&sW[r * 128 + c8], gW + (size_t)r * N + c8);
    }
    wait_asynccnt0();
    __syncthreads();
  }

  const int rowA = blockIdx.x * 128 + wave * 16 + l15;
  v8f acc[8];
#pragma unroll
  for (int t = 0; t < 8; ++t) acc[t] = {};

  const float* xr = X + (size_t)rowA * ldx;
  for (int kk = 0; kk < KPAD; kk += 32) {
    // A fragment (16x32 bf16): lane = row, element e -> K = 16*(e>>3)+8*half+(e&7)
    v16bf a;
#pragma unroll
    for (int e = 0; e < 16; ++e) {
      int k = kk + 16 * (e >> 3) + 8 * half + (e & 7);
      a[e] = f2bf(k < K ? xr[k] : 0.0f);
    }
    // B fragment (32x16 bf16): lane = col, element e -> K = 16*(e>>3)+8*half+(e&7)
    const bf16_t* scol = &sW[kk * 128 + l15];
#pragma unroll
    for (int t = 0; t < 8; ++t) {
      v16bf b;
#pragma unroll
      for (int e = 0; e < 16; ++e)
        b[e] = scol[(16 * (e >> 3) + 8 * half + (e & 7)) * 128 + t * 16];
      acc[t] = __builtin_amdgcn_wmma_f32_16x16x32_bf16(
          false, a, false, b, (short)0, acc[t], false, false);
    }
  }
  // D layout: col = lane&15, row = r + 8*half
  const int rowD = blockIdx.x * 128 + wave * 16 + 8 * half;
#pragma unroll
  for (int t = 0; t < 8; ++t) {
    const int colD = colBase + t * 16 + l15;
#pragma unroll
    for (int r = 0; r < 8; ++r)
      Y[(size_t)(rowD + r) * N + colD] = acc[t][r];
  }
}

// ---------------------------------------------------------------------------
// Degree: deg pre-filled with 1.0 (self loop); add m over dst
// ---------------------------------------------------------------------------
__global__ void k_deg(const int* __restrict__ dst, const float* __restrict__ m,
                      float* __restrict__ deg, int E) {
  int e = blockIdx.x * blockDim.x + threadIdx.x;
  if (e >= E) return;
  float mv = m ? m[e] : 1.0f;
  if (mv != 0.0f) atomAddF(&deg[dst[e]], mv);
}

__global__ void k_rsqrt(const float* __restrict__ deg, float* __restrict__ dinv, int n) {
  int i = blockIdx.x * blockDim.x + threadIdx.x;
  if (i < n) dinv[i] = rsqrtf(deg[i]);
}

// ---------------------------------------------------------------------------
// Edge scatter: agg[dst] += xw[src] * dinv[src]*dinv[dst]*m   (32 lanes / edge)
// ---------------------------------------------------------------------------
__global__ void k_scatter(const float* __restrict__ xw,
                          const int* __restrict__ src, const int* __restrict__ dst,
                          const float* __restrict__ m, const float* __restrict__ dinv,
                          float* __restrict__ agg, int E, int F) {
  int t = blockIdx.x * blockDim.x + threadIdx.x;
  int e = t >> 5, lane = t & 31;
  if (e >= E) return;
  float mv = m ? m[e] : 1.0f;
  if (mv == 0.0f) return;
  int s = src[e], d = dst[e];
  float coef = dinv[s] * dinv[d] * mv;
  const float* xs = xw + (size_t)s * F;
  float* ad = agg + (size_t)d * F;
  for (int f = lane; f < F; f += 32) atomAddF(&ad[f], xs[f] * coef);
}

// ---------------------------------------------------------------------------
// Combine: agg += xw * dinv^2 + b     (self-loop term + bias, in place)
// ---------------------------------------------------------------------------
__global__ void k_combine(float* __restrict__ agg, const float* __restrict__ xw,
                          const float* __restrict__ dinv, const float* __restrict__ b,
                          long total, int F) {
  long i = (long)blockIdx.x * blockDim.x + threadIdx.x;
  if (i >= total) return;
  long n = i / F; int f = (int)(i % F);
  float di = dinv[n];
  agg[i] += xw[i] * di * di + b[f];
}

// ---------------------------------------------------------------------------
// GraphNorm stats: one block per graph, blockDim = F
// var(o) = E[x^2] - 2*a*mu^2 + a^2*mu^2
// ---------------------------------------------------------------------------
__global__ void k_gnorm_stats(const float* __restrict__ x, const float* __restrict__ ga,
                              float* __restrict__ mu, float* __restrict__ rstd,
                              int npg, int F) {
  int g = blockIdx.x, f = threadIdx.x;
  const float* base = x + (size_t)g * npg * F + f;
  float s = 0.f, ss = 0.f;
  for (int n = 0; n < npg; ++n) {
    float v = base[(size_t)n * F];
    s += v; ss += v * v;
  }
  float m = s / npg, a = ga[f];
  float var = ss / npg - 2.f * a * m * m + a * a * m * m;
  mu[g * F + f]   = m;
  rstd[g * F + f] = rsqrtf(var + 1e-5f);
}

// ---------------------------------------------------------------------------
// pk weight inverse norm (one block)
// ---------------------------------------------------------------------------
__global__ void k_pwnorm(const float* __restrict__ pw, float* __restrict__ pwinv, int F) {
  __shared__ float red[256];
  int f = threadIdx.x;
  red[f] = (f < F) ? pw[f] * pw[f] : 0.f;
  __syncthreads();
  for (int s = 128; s > 0; s >>= 1) {
    if (f < s) red[f] += red[f + s];
    __syncthreads();
  }
  if (f == 0) pwinv[0] = rsqrtf(red[0]);
}

// ---------------------------------------------------------------------------
// GraphNorm apply + ReLU + TopK score: one block per node, blockDim = F
// ---------------------------------------------------------------------------
__global__ void k_gnorm_apply_score(
    const float* __restrict__ xin, float* __restrict__ y, float* __restrict__ scores,
    const float* __restrict__ gw, const float* __restrict__ gb, const float* __restrict__ ga,
    const float* __restrict__ mu, const float* __restrict__ rstd,
    const float* __restrict__ pw, const float* __restrict__ pwinv,
    int npg, int F) {
  __shared__ float red[256];
  int node = blockIdx.x, f = threadIdx.x;
  int g = node / npg;
  float v = xin[(size_t)node * F + f];
  float o = v - ga[f] * mu[g * F + f];
  float r = gw[f] * o * rstd[g * F + f] + gb[f];
  r = fmaxf(r, 0.0f);                        // ReLU (dropout = identity, eval)
  y[(size_t)node * F + f] = r;
  red[f] = r * pw[f];
  __syncthreads();
  for (int s = F >> 1; s > 0; s >>= 1) {
    if (f < s) red[f] += red[f + s];
    __syncthreads();
  }
  if (f == 0) scores[node] = tanhf(red[0] * pwinv[0]);
}

// ---------------------------------------------------------------------------
// TopK: per-graph bitonic sort (descending, lower index wins ties) in LDS
// ---------------------------------------------------------------------------
__global__ void __launch_bounds__(1024) k_topk(
    const float* __restrict__ scores, int* __restrict__ perm, float* __restrict__ vals,
    int npg, int ksel) {
  __shared__ float sv[4096];
  __shared__ int   si[4096];
  int g = blockIdx.x, tid = threadIdx.x, bd = blockDim.x;
  for (int i = tid; i < npg; i += bd) { sv[i] = scores[g * npg + i]; si[i] = i; }
  __syncthreads();
  for (int k2 = 2; k2 <= npg; k2 <<= 1) {
    for (int j = k2 >> 1; j > 0; j >>= 1) {
      for (int i = tid; i < npg; i += bd) {
        int ixj = i ^ j;
        if (ixj > i) {
          float v1 = sv[i], v2 = sv[ixj];
          int i1 = si[i], i2 = si[ixj];
          bool gt = (v1 > v2) || (v1 == v2 && i1 < i2);
          bool up = ((i & k2) == 0);          // up-blocks sorted descending
          if (up ? !gt : gt) {
            sv[i] = v2; sv[ixj] = v1; si[i] = i2; si[ixj] = i1;
          }
        }
      }
      __syncthreads();
    }
  }
  for (int i = tid; i < ksel; i += bd) {
    perm[g * ksel + i] = g * npg + si[i];     // global old node id
    vals[g * ksel + i] = sv[i];
  }
}

__global__ void k_set_mapping(const int* __restrict__ perm, int* __restrict__ mapping,
                              int total) {
  int i = blockIdx.x * blockDim.x + threadIdx.x;
  if (i < total) mapping[perm[i]] = i;        // new global id
}

// Gate kept nodes: out[j] = x[perm[j]] * vals[j]
__global__ void k_gate(const float* __restrict__ x, const int* __restrict__ perm,
                       const float* __restrict__ vals, float* __restrict__ out,
                       long total, int F) {
  long i = (long)blockIdx.x * blockDim.x + threadIdx.x;
  if (i >= total) return;
  long j = i / F; int f = (int)(i % F);
  out[i] = x[(size_t)perm[j] * F + f] * vals[j];
}

__global__ void k_remap(const int* __restrict__ src, const int* __restrict__ dst,
                        const float* __restrict__ m, const int* __restrict__ mapping,
                        int* __restrict__ nsrc, int* __restrict__ ndst,
                        float* __restrict__ nm, int E) {
  int e = blockIdx.x * blockDim.x + threadIdx.x;
  if (e >= E) return;
  int s = mapping[src[e]], d = mapping[dst[e]];
  float mv = m ? m[e] : 1.0f;
  bool ok = (s >= 0) && (d >= 0);
  nsrc[e] = ok ? s : 0;
  ndst[e] = ok ? d : 0;
  nm[e]   = ok ? mv : 0.0f;
}

// ---------------------------------------------------------------------------
// Final graph pooling: mean + max over 512 nodes, accumulated over timesteps
// ---------------------------------------------------------------------------
__global__ void k_pool(const float* __restrict__ x, float* __restrict__ outacc,
                       int npg, int F) {
  int g = blockIdx.x, f = threadIdx.x;
  const float* base = x + (size_t)g * npg * F + f;
  float s = 0.f, mx = -3.402823466e38f;
  for (int n = 0; n < npg; ++n) {
    float v = base[(size_t)n * F];
    s += v; mx = fmaxf(mx, v);
  }
  outacc[g * 2 * F + f]     += s / npg;
  outacc[g * 2 * F + F + f] += mx;
}

__global__ void k_finalize(const float* __restrict__ acc, float* __restrict__ out,
                           int n, float scale) {
  int i = blockIdx.x * blockDim.x + threadIdx.x;
  if (i < n) out[i] = acc[i] * scale;
}

// ---------------------------------------------------------------------------
// Host orchestration
// ---------------------------------------------------------------------------
static inline long cdivl(long a, long b) { return (a + b - 1) / b; }

extern "C" void kernel_launch(void* const* d_in, const int* in_sizes, int n_in,
                              void* d_out, int out_size, void* d_ws, size_t ws_size,
                              hipStream_t stream) {
  (void)in_sizes; (void)n_in; (void)out_size; (void)ws_size;

  const float* x_seq = (const float*)d_in[0];
  const int*   eidx  = (const int*)d_in[1];
  const float* Wf[3]  = {(const float*)d_in[2], (const float*)d_in[4], (const float*)d_in[6]};
  const float* bsp[3] = {(const float*)d_in[3], (const float*)d_in[5], (const float*)d_in[7]};
  const float* gwp[3] = {(const float*)d_in[8],  (const float*)d_in[11], (const float*)d_in[14]};
  const float* gbp[3] = {(const float*)d_in[9],  (const float*)d_in[12], (const float*)d_in[15]};
  const float* gap[3] = {(const float*)d_in[10], (const float*)d_in[13], (const float*)d_in[16]};
  const float* pwp[3] = {(const float*)d_in[17], (const float*)d_in[18], (const float*)d_in[19]};

  const int Kdim[3] = {F_IN, 128, 128};
  const int Fdim[3] = {128, 128, OUTF};

  // ---- workspace carve -----------------------------------------------------
  char* w = (char*)d_ws;
  auto carve = [&](size_t bytes) {
    void* p = (void*)w;
    w += (bytes + 255) & ~(size_t)255;
    return p;
  };
  float* XW    = (float*)carve((size_t)NB * NPG0 * 128 * 4);   // 32 MB (also holds y)
  float* AGG   = (float*)carve((size_t)NB * NPG0 * 128 * 4);   // 32 MB
  float* BUFX  = (float*)carve((size_t)NB * 2048 * 128 * 4);   // 16 MB (gated x)
  float* DEG   = (float*)carve((size_t)NB * NPG0 * 4);
  float* DINV  = (float*)carve((size_t)NB * NPG0 * 4);
  float* SCOR  = (float*)carve((size_t)NB * NPG0 * 4);
  int*   MAPP  = (int*)  carve((size_t)NB * NPG0 * 4);
  int*   PERM  = (int*)  carve((size_t)NB * 2048 * 4);
  float* VALS  = (float*)carve((size_t)NB * 2048 * 4);
  float* MU    = (float*)carve((size_t)NB * OUTF * 4);
  float* RSTD  = (float*)carve((size_t)NB * OUTF * 4);
  float* PWINV = (float*)carve(256);
  int*   ES[2]; int* ED[2]; float* EM[2];
  for (int i = 0; i < 2; ++i) {
    ES[i] = (int*)  carve((size_t)EDGES * 4);
    ED[i] = (int*)  carve((size_t)EDGES * 4);
    EM[i] = (float*)carve((size_t)EDGES * 4);
  }
  bf16_t* WB[3];
  WB[0] = (bf16_t*)carve((size_t)KPAD * 128 * 2);
  WB[1] = (bf16_t*)carve((size_t)KPAD * 128 * 2);
  WB[2] = (bf16_t*)carve((size_t)KPAD * OUTF * 2);
  float* OUTACC = (float*)carve((size_t)NB * 2 * OUTF * 4);

  // ---- weight conversion (bf16, K zero-padded to KPAD) ---------------------
  for (int L = 0; L < 3; ++L) {
    int tot = KPAD * Fdim[L];
    k_conv_w<<<(unsigned)cdivl(tot, 256), 256, 0, stream>>>(Wf[L], WB[L], Kdim[L], KPAD, Fdim[L]);
  }
  k_fill_f32<<<(unsigned)cdivl(NB * 2 * OUTF, 256), 256, 0, stream>>>(OUTACC, 0.f, NB * 2 * OUTF);

  // ---- timestep loop -------------------------------------------------------
  for (int t = 0; t < NT; ++t) {
    const float* xptr = x_seq + (size_t)t * NB * NPG0 * F_IN;
    int ldx = F_IN;
    const int*   csrc = eidx + (size_t)t * 2 * EDGES;
    const int*   cdst = csrc + EDGES;
    const float* cm   = nullptr;
    int nodes = NB * NPG0;
    int npg   = NPG0;

    for (int L = 0; L < 3; ++L) {
      const int K = Kdim[L], F = Fdim[L];
      const long totNF = (long)nodes * F;

      // 1) xw = x @ W  (WMMA bf16, weights async-staged in LDS)
      dim3 gg(nodes / 128, F / 128);
      k_gemm_bf16<<<gg, 256, 0, stream>>>(xptr, ldx, K, WB[L], F, XW);

      // 2) degrees + dinv
      k_fill_f32<<<(unsigned)cdivl(nodes, 256), 256, 0, stream>>>(DEG, 1.0f, nodes);
      k_deg<<<(unsigned)cdivl(EDGES, 256), 256, 0, stream>>>(cdst, cm, DEG, EDGES);
      k_rsqrt<<<(unsigned)cdivl(nodes, 256), 256, 0, stream>>>(DEG, DINV, nodes);

      // 3) edge aggregation
      k_fill_f32<<<(unsigned)cdivl(totNF, 256), 256, 0, stream>>>(AGG, 0.f, totNF);
      k_scatter<<<(unsigned)cdivl((long)EDGES * 32, 256), 256, 0, stream>>>(
          XW, csrc, cdst, cm, DINV, AGG, EDGES, F);

      // 4) self-loop term + bias (into AGG)
      k_combine<<<(unsigned)cdivl(totNF, 256), 256, 0, stream>>>(AGG, XW, DINV, bsp[L], totNF, F);

      // 5) GraphNorm + ReLU + TopK score (y written into XW)
      k_gnorm_stats<<<NB, F, 0, stream>>>(AGG, gap[L], MU, RSTD, npg, F);
      k_pwnorm<<<1, 256, 0, stream>>>(pwp[L], PWINV, F);
      k_gnorm_apply_score<<<nodes, F, 0, stream>>>(
          AGG, XW, SCOR, gwp[L], gbp[L], gap[L], MU, RSTD, pwp[L], PWINV, npg, F);

      // 6) TopK pooling
      const int ksel = npg >> 1;
      k_topk<<<NB, 1024, 0, stream>>>(SCOR, PERM, VALS, npg, ksel);
      k_fill_i32<<<(unsigned)cdivl(nodes, 256), 256, 0, stream>>>(MAPP, -1, nodes);
      k_set_mapping<<<(unsigned)cdivl(NB * ksel, 256), 256, 0, stream>>>(PERM, MAPP, NB * ksel);
      k_gate<<<(unsigned)cdivl((long)NB * ksel * F, 256), 256, 0, stream>>>(
          XW, PERM, VALS, BUFX, (long)NB * ksel * F, F);
      if (L < 2) {
        k_remap<<<(unsigned)cdivl(EDGES, 256), 256, 0, stream>>>(
            csrc, cdst, cm, MAPP, ES[L], ED[L], EM[L], EDGES);
        csrc = ES[L]; cdst = ED[L]; cm = EM[L];
      }

      xptr = BUFX; ldx = F; nodes = NB * ksel; npg = ksel;
    }

    // 7) mean/max pooling over 512 kept nodes, accumulate over timesteps
    k_pool<<<NB, OUTF, 0, stream>>>(BUFX, OUTACC, npg, OUTF);
  }

  // temporal mean
  k_finalize<<<(unsigned)cdivl(NB * 2 * OUTF, 256), 256, 0, stream>>>(
      OUTACC, (float*)d_out, NB * 2 * OUTF, 1.0f / NT);
}
